// CodecT_44452911514177
// MI455X (gfx1250) — compile-verified, matrix operand order checked by reading, not run on hardware
//
#include <hip/hip_runtime.h>
#include <math.h>

// ---------------------------------------------------------------------------
// Types / constants
// ---------------------------------------------------------------------------
typedef _Float16 h16;
typedef h16   v8h  __attribute__((ext_vector_type(8)));
typedef h16   v16h __attribute__((ext_vector_type(16)));
typedef float v8f  __attribute__((ext_vector_type(8)));

static constexpr int BB   = 8;
static constexpr int TT   = 800;
static constexpr int INDIM= 80;
static constexpr int LAT  = 256;
static constexpr int PDIM = 64;
static constexpr int MROWS= BB * TT;     // 6400
static constexpr float LRELU_SLOPE = 0.01f;

#define ASYNC_LDS 1   // gfx1250 global_load_async_to_lds_b128 staging

typedef __attribute__((address_space(3))) const h16 as3_ch16;
static __device__ __forceinline__ unsigned lds_off(const h16* p) {
  // addrspace(3) ptrtoint == raw LDS byte offset (what VDST of async ops wants)
  return (unsigned)(unsigned long long)(as3_ch16*)p;
}
static __device__ __forceinline__ void async_b128(const h16* lds, const h16* g) {
#if ASYNC_LDS
  asm volatile("global_load_async_to_lds_b128 %0, %1, off"
               :: "v"(lds_off(lds)), "v"(g) : "memory");
#endif
}
// Wait until ASYNCcnt <= n (n = #asyncs issued for the in-flight next tile).
// ASYNCcnt decrements in order, so this guarantees the *previous* tile-group
// has fully landed in LDS while the next group stays outstanding.
static __device__ __forceinline__ void wait_async_n(int n) {
#if ASYNC_LDS
  if (n >= 3)      asm volatile("s_wait_asynccnt 0x3" ::: "memory");
  else if (n == 2) asm volatile("s_wait_asynccnt 0x2" ::: "memory");
  else if (n == 1) asm volatile("s_wait_asynccnt 0x1" ::: "memory");
  else             asm volatile("s_wait_asynccnt 0x0" ::: "memory");
#endif
}

// ---------------------------------------------------------------------------
// WMMA GEMM:  D = act(alpha * A @ B^T + bias) + res
//   A: f16 [M,K] row-major (lda);  B: f16 [N,K] row-major (ldb)
//   All K and N used by this model are multiples of 8 -> 8-wide loads only.
// Batched: z -> (zb=z/H, zh=z%H); base = zb*sb + zh*sh per operand.
// Block: 128x64 C-tile, 256 thr = 8 waves (4x2), wave = 32x32 -> 4 WMMA/K-step.
// Staging: double-buffered LDS; async global->LDS b128 pipelined one tile
// ahead for full interior tiles, register path for edge tiles.
// ---------------------------------------------------------------------------
struct GemmP {
  int  M, N, K;
  int  lda, ldb, ldd;
  long A_sb, A_sh, B_sb, B_sh, D_sb, D_sh;
  int  H;
  float alpha;
  int  act;     // 0 = none, 1 = leaky_relu(0.01)
};

__global__ __launch_bounds__(256) void k_gemm(const h16* __restrict__ A,
                                              const h16* __restrict__ Bm,
                                              const float* __restrict__ bias,
                                              const float* __restrict__ res,
                                              float* __restrict__ Dm, GemmP p) {
  __shared__ h16 sA[2][128][40];   // 40 = 32 + pad, rows stay 16B aligned
  __shared__ h16 sB[2][64][40];    // stored [n][k]

  const int z  = blockIdx.z;
  const int zb = z / p.H, zh = z % p.H;
  const long baseA = (long)zb * p.A_sb + (long)zh * p.A_sh;
  const long baseB = (long)zb * p.B_sb + (long)zh * p.B_sh;
  const long baseD = (long)zb * p.D_sb + (long)zh * p.D_sh;

  const int m0 = blockIdx.y * 128;
  const int n0 = blockIdx.x * 64;
  const int t    = threadIdx.x;
  const int wave = t >> 5;
  const int lane = t & 31;
  const int wm = (wave >> 1) * 32;   // 0,32,64,96
  const int wn = (wave & 1) * 32;    // 0,32
  const int lr = lane & 15;
  const int kb = (lane < 16) ? 0 : 8;  // 16-bit operand lane K-base (ISA layout)

  const int ra = t >> 2, ca = (t & 3) * 8;   // staging coords

  v8f c00 = {}, c01 = {}, c10 = {}, c11 = {};

  // Stage one 128x32 A-tile + 64x32 B-tile into buffer `buf`; returns #asyncs.
  auto stageTile = [&](int k0, int buf) -> int {
    int cnt = 0;
    const bool kFull = (k0 + 32 <= p.K);
    const bool aFull = (m0 + 128 <= p.M) && kFull;
    const bool bFull = (n0 + 64 <= p.N) && kFull;
    if (aFull) {
      async_b128(&sA[buf][ra][ca],      A + baseA + (long)(m0 + ra) * p.lda + (k0 + ca));
      async_b128(&sA[buf][ra + 64][ca], A + baseA + (long)(m0 + ra + 64) * p.lda + (k0 + ca));
      cnt += 2;
    } else {
#pragma unroll
      for (int q = 0; q < 2; ++q) {
        const int r = ra + q * 64;
        const int gm = m0 + r, gk = k0 + ca;
        v8h v = {};
        if (gm < p.M && gk + 8 <= p.K)
          v = *(const v8h*)(A + baseA + (long)gm * p.lda + gk);
        *(v8h*)&sA[buf][r][ca] = v;
      }
    }
    if (bFull) {
      async_b128(&sB[buf][ra][ca], Bm + baseB + (long)(n0 + ra) * p.ldb + (k0 + ca));
      cnt += 1;
    } else {
      const int gn = n0 + ra, gk = k0 + ca;
      v8h v = {};
      if (gn < p.N && gk + 8 <= p.K)
        v = *(const v8h*)(Bm + baseB + (long)gn * p.ldb + gk);
      *(v8h*)&sB[buf][ra][ca] = v;
      // next-tile cache prefetch on the edge path (global_prefetch_b8)
      if (k0 + 32 < p.K && gn < p.N)
        __builtin_prefetch(Bm + baseB + (long)gn * p.ldb + (k0 + 32), 0, 3);
    }
    return cnt;
  };

  const int niter = (p.K + 31) >> 5;
  (void)stageTile(0, 0);

  for (int it = 0; it < niter; ++it) {
    int nc = 0;
    if (it + 1 < niter) nc = stageTile((it + 1) << 5, (it + 1) & 1);
    wait_async_n(nc);
    __syncthreads();
    const int buf = it & 1;

    v16h a0, a1, b0, b1;
    {
      v8h lo = *(const v8h*)&sA[buf][wm + lr][kb];
      v8h hi = *(const v8h*)&sA[buf][wm + lr][kb + 16];
#pragma unroll
      for (int i = 0; i < 8; ++i) { a0[i] = lo[i]; a0[i + 8] = hi[i]; }
      lo = *(const v8h*)&sA[buf][wm + 16 + lr][kb];
      hi = *(const v8h*)&sA[buf][wm + 16 + lr][kb + 16];
#pragma unroll
      for (int i = 0; i < 8; ++i) { a1[i] = lo[i]; a1[i + 8] = hi[i]; }
    }
    {
      v8h lo = *(const v8h*)&sB[buf][wn + lr][kb];
      v8h hi = *(const v8h*)&sB[buf][wn + lr][kb + 16];
#pragma unroll
      for (int i = 0; i < 8; ++i) { b0[i] = lo[i]; b0[i + 8] = hi[i]; }
      lo = *(const v8h*)&sB[buf][wn + 16 + lr][kb];
      hi = *(const v8h*)&sB[buf][wn + 16 + lr][kb + 16];
#pragma unroll
      for (int i = 0; i < 8; ++i) { b1[i] = lo[i]; b1[i + 8] = hi[i]; }
    }
    c00 = __builtin_amdgcn_wmma_f32_16x16x32_f16(false, a0, false, b0, (short)0, c00, false, false);
    c01 = __builtin_amdgcn_wmma_f32_16x16x32_f16(false, a0, false, b1, (short)0, c01, false, false);
    c10 = __builtin_amdgcn_wmma_f32_16x16x32_f16(false, a1, false, b0, (short)0, c10, false, false);
    c11 = __builtin_amdgcn_wmma_f32_16x16x32_f16(false, a1, false, b1, (short)0, c11, false, false);
    __syncthreads();
  }

  // ---- epilogue: C layout lane l, vgpr r -> (M = r + 8*(l>=16), N = l&15) ----
  const int mofs = (lane >> 4) * 8;
#pragma unroll
  for (int ai = 0; ai < 2; ++ai) {
#pragma unroll
    for (int j = 0; j < 2; ++j) {
      v8f& cc = ai ? (j ? c11 : c10) : (j ? c01 : c00);
      const int gn = n0 + wn + j * 16 + lr;
      if (gn >= p.N) continue;
      const float bv = bias ? bias[gn] : 0.f;
#pragma unroll
      for (int r = 0; r < 8; ++r) {
        const int gm = m0 + wm + ai * 16 + mofs + r;
        if (gm >= p.M) continue;
        float v = cc[r] * p.alpha + bv;
        if (p.act == 1) v = v > 0.f ? v : LRELU_SLOPE * v;
        const long di = baseD + (long)gm * p.ldd + gn;
        if (res) v += res[di];
        Dm[di] = v;
      }
    }
  }
}

// ---------------------------------------------------------------------------
// Support kernels
// ---------------------------------------------------------------------------
__global__ void k_cvt(const float* __restrict__ a, h16* __restrict__ o, long n) {
  long i = (long)blockIdx.x * 256 + threadIdx.x;
  if (i < n) o[i] = (h16)a[i];
}

// a[R,C] f32 -> o[C,R] f16   (weight pre-transposition so GEMM B is [N,K])
__global__ void k_cvt_t(const float* __restrict__ a, h16* __restrict__ o,
                        int R, int C) {
  long i = (long)blockIdx.x * 256 + threadIdx.x;
  const long total = (long)R * C;
  if (i >= total) return;
  const int cc = (int)(i % C);
  const long r = i / C;
  o[(long)cc * R + r] = (h16)a[i];
}

// V f32 [B*T, H*hd] -> Vt f16 [B, H, hd, T]
__global__ void k_vt(const float* __restrict__ v, h16* __restrict__ o,
                     int H, int hd) {
  const int d = H * hd;
  long i = (long)blockIdx.x * 256 + threadIdx.x;
  const long total = (long)MROWS * d;
  if (i >= total) return;
  const int e = (int)(i % d);
  const long bt = i / d;
  const int tt = (int)(bt % TT);
  const int bidx = (int)(bt / TT);
  const int h = e / hd, ee = e % hd;
  o[(((long)(bidx * H + h)) * hd + ee) * TT + tt] = (h16)v[i];
}

__global__ void k_ln(const float* __restrict__ x, float* __restrict__ y,
                     const float* __restrict__ s, const float* __restrict__ b,
                     int rows, int D) {
  const int wave = threadIdx.x >> 5, lane = threadIdx.x & 31;
  const int row = blockIdx.x * (blockDim.x >> 5) + wave;
  if (row >= rows) return;
  const float* xr = x + (long)row * D;
  float sum = 0.f, ss = 0.f;
  for (int j = lane; j < D; j += 32) { float v = xr[j]; sum += v; ss += v * v; }
  for (int m = 16; m; m >>= 1) {
    sum += __shfl_xor(sum, m, 32);
    ss  += __shfl_xor(ss, m, 32);
  }
  const float mu  = sum / D;
  const float var = ss / D - mu * mu;
  const float rs  = rsqrtf(var + 1e-5f);
  float* yr = y + (long)row * D;
  for (int j = lane; j < D; j += 32) yr[j] = (xr[j] - mu) * rs * s[j] + b[j];
}

__global__ void k_softmax(float* __restrict__ x, int rows, int L) {
  __shared__ float red[256];
  const int row = blockIdx.x;
  if (row >= rows) return;
  float* xr = x + (long)row * L;
  float mx = -3.4e38f;
  for (int j = threadIdx.x; j < L; j += 256) mx = fmaxf(mx, xr[j]);
  red[threadIdx.x] = mx; __syncthreads();
  for (int s = 128; s; s >>= 1) {
    if (threadIdx.x < s) red[threadIdx.x] = fmaxf(red[threadIdx.x], red[threadIdx.x + s]);
    __syncthreads();
  }
  mx = red[0]; __syncthreads();
  float sum = 0.f;
  for (int j = threadIdx.x; j < L; j += 256) {
    float e = __expf(xr[j] - mx);
    xr[j] = e; sum += e;
  }
  red[threadIdx.x] = sum; __syncthreads();
  for (int s = 128; s; s >>= 1) {
    if (threadIdx.x < s) red[threadIdx.x] += red[threadIdx.x + s];
    __syncthreads();
  }
  const float inv = 1.f / red[0];
  for (int j = threadIdx.x; j < L; j += 256) xr[j] *= inv;
}

__global__ void k_rownorm(const float* __restrict__ cb, float* __restrict__ cn,
                          int Kc, int D) {
  const int wave = threadIdx.x >> 5, lane = threadIdx.x & 31;
  const int r = blockIdx.x * (blockDim.x >> 5) + wave;
  if (r >= Kc) return;
  float s = 0.f;
  for (int j = lane; j < D; j += 32) { float v = cb[(long)r * D + j]; s += v * v; }
  for (int m = 16; m; m >>= 1) s += __shfl_xor(s, m, 32);
  if (lane == 0) cn[r] = s;
}

__global__ void k_argmin_gather(const float* __restrict__ S,
                                const float* __restrict__ cn,
                                const float* __restrict__ cb,
                                float* __restrict__ q, int Kc, int D) {
  __shared__ float bs[256];
  __shared__ int   bi[256];
  const long row = blockIdx.x;
  float best = 3.4e38f; int bidx = 0;
  for (int j = threadIdx.x; j < Kc; j += 256) {
    float v = S[row * Kc + j] + cn[j];
    if (v < best) { best = v; bidx = j; }
  }
  bs[threadIdx.x] = best; bi[threadIdx.x] = bidx; __syncthreads();
  for (int s = 128; s; s >>= 1) {
    if (threadIdx.x < s && bs[threadIdx.x + s] < bs[threadIdx.x]) {
      bs[threadIdx.x] = bs[threadIdx.x + s];
      bi[threadIdx.x] = bi[threadIdx.x + s];
    }
    __syncthreads();
  }
  const int idx = bi[0];
  for (int d = threadIdx.x; d < D; d += 256)
    q[row * D + d] = cb[(long)idx * D + d];
}

// Conv1d(k=3, stride=2, pad=1), NTC layout; optional leaky-relu on inputs.
__global__ void k_conv_ds(const float* __restrict__ x, const float* __restrict__ w,
                          const float* __restrict__ b, float* __restrict__ y,
                          int Bc, int Tin, int D, int lrelu_in) {
  const long i = (long)blockIdx.x * 256 + threadIdx.x;
  const int Tout = Tin >> 1;
  const long total = (long)Bc * Tout * D;
  if (i >= total) return;
  const int o = (int)(i % D);
  long tmp = i / D;
  const int tt = (int)(tmp % Tout);
  const int bb = (int)(tmp / Tout);
  float acc = b[o];
  const float* xb = x + (long)bb * Tin * D;
  for (int kk = 0; kk < 3; ++kk) {
    const int ti = 2 * tt - 1 + kk;
    if (ti < 0 || ti >= Tin) continue;
    const float* xr = xb + (long)ti * D;
    const float* wr = w + ((long)o * D) * 3 + kk;  // w is (O, I, 3)
    for (int ii = 0; ii < D; ++ii) {
      float v = xr[ii];
      if (lrelu_in) v = v > 0.f ? v : LRELU_SLOPE * v;
      acc += v * wr[ii * 3];
    }
  }
  y[i] = acc;
}

__global__ void k_repeat(const float* __restrict__ in, float* __restrict__ out,
                         int Bc, int Tout, int D, int rate) {
  const long i = (long)blockIdx.x * 256 + threadIdx.x;
  const long total = (long)Bc * Tout * D;
  if (i >= total) return;
  const int dd = (int)(i % D);
  long tmp = i / D;
  const int t = (int)(tmp % Tout);
  const int bb = (int)(tmp / Tout);
  out[i] = in[(((long)bb * (Tout / rate)) + t / rate) * D + dd];
}

__global__ void k_add(const float* a, const float* b, float* o, long n) {
  long i = (long)blockIdx.x * 256 + threadIdx.x;
  if (i < n) o[i] = a[i] + b[i];
}
__global__ void k_sub(const float* a, const float* b, float* o, long n) {
  long i = (long)blockIdx.x * 256 + threadIdx.x;
  if (i < n) o[i] = a[i] - b[i];
}
__global__ void k_concat3(const float* a, int Da, const float* b, int Db,
                          const float* c, int Dc, float* o, long rows) {
  const long i = (long)blockIdx.x * 256 + threadIdx.x;
  const int Dt = Da + Db + Dc;
  const long total = rows * Dt;
  if (i >= total) return;
  const long r = i / Dt;
  const int d = (int)(i % Dt);
  float v;
  if (d < Da)            v = a[r * Da + d];
  else if (d < Da + Db)  v = b[r * Db + (d - Da)];
  else                   v = c[r * Dc + (d - Da - Db)];
  o[i] = v;
}

// ---------------------------------------------------------------------------
// Host-side orchestration
// ---------------------------------------------------------------------------
namespace {

struct Ctx {
  hipStream_t s;
  char* base; size_t off, cap;
  void* alloc(size_t bytes) {
    size_t o = (off + 255) & ~size_t(255);
    off = o + bytes;
    return base + o;
  }
  float* f32(long n) { return (float*)alloc((size_t)n * 4); }
  h16*   f16(long n) { return (h16*)alloc((size_t)n * 2); }
};
struct ASave {
  Ctx& c; size_t o;
  explicit ASave(Ctx& cc) : c(cc), o(cc.off) {}
  ~ASave() { c.off = o; }
};

inline long ceil_div(long a, long b) { return (a + b - 1) / b; }

h16* cvt(Ctx& c, const float* a, long n) {
  h16* o = c.f16(n);
  k_cvt<<<dim3((unsigned)ceil_div(n, 256)), 256, 0, c.s>>>(a, o, n);
  return o;
}
h16* cvtT(Ctx& c, const float* a, int R, int C) {  // -> f16 [C,R]
  h16* o = c.f16((long)R * C);
  k_cvt_t<<<dim3((unsigned)ceil_div((long)R * C, 256)), 256, 0, c.s>>>(a, o, R, C);
  return o;
}

// D = act(alpha * A[M,K] @ B[N,K]^T + bias) + res
void gemm(Ctx& c, const h16* A, const h16* B, const float* bias,
          const float* res, float* D, int M, int N, int K, int lda, int ldb,
          int ldd, float alpha = 1.f, int act = 0, int batch = 1, int H = 1,
          long A_sb = 0, long A_sh = 0, long B_sb = 0, long B_sh = 0,
          long D_sb = 0, long D_sh = 0) {
  GemmP p;
  p.M = M; p.N = N; p.K = K; p.lda = lda; p.ldb = ldb; p.ldd = ldd;
  p.A_sb = A_sb; p.A_sh = A_sh; p.B_sb = B_sb; p.B_sh = B_sh;
  p.D_sb = D_sb; p.D_sh = D_sh; p.H = H; p.alpha = alpha; p.act = act;
  dim3 g((unsigned)ceil_div(N, 64), (unsigned)ceil_div(M, 128), (unsigned)batch);
  k_gemm<<<g, 256, 0, c.s>>>(A, B, bias, res, D, p);
}

void ln(Ctx& c, const float* x, float* y, const float* s, const float* b,
        int rows, int D) {
  k_ln<<<dim3((unsigned)ceil_div(rows, 8)), 256, 0, c.s>>>(x, y, s, b, rows, D);
}

// ---- parameter structs, loaded from d_in in JAX sorted-pytree order ----
struct Cur {
  void* const* in; int i;
  const float* next() { return (const float*)in[i++]; }
};
struct Lin  { const float* b; const h16* wt; int din, dout; };      // wt: [dout,din]
struct MHAW { const h16 *wk, *wo, *wq, *wv; };                      // all [dout,din]
struct FFW  { const float *b1, *b2; const h16 *w1, *w2; };          // transposed
struct ConvW{ const float* b; const float* w; };
struct EncP {
  MHAW attn; ConvW ds[2]; int nds; FFW ff;
  const float *ln1b, *ln1s, *ln2b, *ln2s;
};
struct DecP {
  MHAW cattn; FFW ff;
  const float *ln1b, *ln1s, *ln2b, *ln2s, *ln3b, *ln3s;
  MHAW sattn;
};
struct CB { const float* cb; const h16* h; float* cn; int Kc, D; };

MHAW loadMHA(Cur& cu, Ctx& x, int d) {      // sorted: wk, wo, wq, wv
  MHAW m;
  m.wk = cvtT(x, cu.next(), d, d);
  m.wo = cvtT(x, cu.next(), d, d);
  m.wq = cvtT(x, cu.next(), d, d);
  m.wv = cvtT(x, cu.next(), d, d);
  return m;
}
FFW loadFF(Cur& cu, Ctx& x, int d) {        // sorted: b1, b2, w1, w2
  FFW f;
  f.b1 = cu.next(); f.b2 = cu.next();
  f.w1 = cvtT(x, cu.next(), d, 2 * d);      // -> [2d, d]
  f.w2 = cvtT(x, cu.next(), 2 * d, d);      // -> [d, 2d]
  return f;
}
EncP loadEnc(Cur& cu, Ctx& x, int d, int nds) {
  EncP e;
  e.attn = loadMHA(cu, x, d);
  e.nds = nds;
  for (int i = 0; i < nds; ++i) { e.ds[i].b = cu.next(); e.ds[i].w = cu.next(); }
  e.ff = loadFF(cu, x, d);
  e.ln1b = cu.next(); e.ln1s = cu.next();
  e.ln2b = cu.next(); e.ln2s = cu.next();
  return e;
}
DecP loadDec(Cur& cu, Ctx& x, int d) {
  DecP p;
  p.cattn = loadMHA(cu, x, d);
  p.ff = loadFF(cu, x, d);
  p.ln1b = cu.next(); p.ln1s = cu.next();
  p.ln2b = cu.next(); p.ln2s = cu.next();
  p.ln3b = cu.next(); p.ln3s = cu.next();
  p.sattn = loadMHA(cu, x, d);
  return p;
}
Lin loadLin(Cur& cu, Ctx& x, int din, int dout) {  // b, w
  Lin l;
  l.b = cu.next();
  l.wt = cvtT(x, cu.next(), din, dout);
  l.din = din; l.dout = dout;
  return l;
}
CB loadCB(Ctx& x, const float* cb, int Kc, int D) {
  CB r;
  r.cb = cb; r.Kc = Kc; r.D = D;
  r.h = cvt(x, cb, (long)Kc * D);  // already [N=Kc, K=D]
  r.cn = x.f32(Kc);
  k_rownorm<<<dim3((unsigned)ceil_div(Kc, 8)), 256, 0, x.s>>>(cb, r.cn, Kc, D);
  return r;
}

// ---- model building blocks (all at B=8, T=800) ----
void mha(Ctx& c, const float* xq, const float* xkv, const MHAW& w, int d,
         int heads, const float* res, float* out) {
  ASave sv(c);
  const int hd = d / heads;
  h16* hq = cvt(c, xq, (long)MROWS * d);
  h16* hk = (xkv == xq) ? hq : cvt(c, xkv, (long)MROWS * d);

  float* Q = c.f32((long)MROWS * d);
  float* K = c.f32((long)MROWS * d);
  float* V = c.f32((long)MROWS * d);
  gemm(c, hq, w.wq, nullptr, nullptr, Q, MROWS, d, d, d, d, d);
  gemm(c, hk, w.wk, nullptr, nullptr, K, MROWS, d, d, d, d, d);
  gemm(c, hk, w.wv, nullptr, nullptr, V, MROWS, d, d, d, d, d);
  h16* Qh = cvt(c, Q, (long)MROWS * d);
  h16* Kh = cvt(c, K, (long)MROWS * d);
  h16* Vt = c.f16((long)MROWS * d);   // [B,H,hd,T]
  k_vt<<<dim3((unsigned)ceil_div((long)MROWS * d, 256)), 256, 0, c.s>>>(V, Vt, heads, hd);

  // scores[z=b*H+h] = (Q_bh @ K_bh^T) / sqrt(hd);  K is [N=T, K=hd] directly
  float* S = c.f32((long)BB * heads * TT * TT);
  gemm(c, Qh, Kh, nullptr, nullptr, S, TT, TT, hd, d, d, TT,
       1.f / sqrtf((float)hd), 0, BB * heads, heads,
       (long)TT * d, hd, (long)TT * d, hd, (long)heads * TT * TT, (long)TT * TT);
  k_softmax<<<dim3((unsigned)(BB * heads * TT)), 256, 0, c.s>>>(S, BB * heads * TT, TT);
  h16* Sh = cvt(c, S, (long)BB * heads * TT * TT);

  // O_bh = P_bh @ V_bh ; Vt is [hd, T] per (b,h) -> B operand [N=hd, K=T]
  float* AO = c.f32((long)MROWS * d);
  gemm(c, Sh, Vt, nullptr, nullptr, AO, TT, hd, TT, TT, TT, d, 1.f, 0,
       BB * heads, heads, (long)heads * TT * TT, (long)TT * TT,
       (long)d * TT, (long)hd * TT, (long)TT * d, hd);
  h16* AOh = cvt(c, AO, (long)MROWS * d);
  gemm(c, AOh, w.wo, nullptr, res, out, MROWS, d, d, d, d, d);
}

void ffn(Ctx& c, const float* x, const FFW& f, int d, const float* res,
         float* out) {
  ASave sv(c);
  h16* hx = cvt(c, x, (long)MROWS * d);
  float* t = c.f32((long)MROWS * 2 * d);
  gemm(c, hx, f.w1, f.b1, nullptr, t, MROWS, 2 * d, d, d, d, 2 * d, 1.f, 1);
  h16* th = cvt(c, t, (long)MROWS * 2 * d);
  gemm(c, th, f.w2, f.b2, res, out, MROWS, d, 2 * d, 2 * d, 2 * d, d);
}

void enc_core(Ctx& c, const float* x, const EncP& p, int d, int heads,
              float* out) {
  ASave sv(c);
  float* h1 = c.f32((long)MROWS * d);
  ln(c, x, h1, p.ln1s, p.ln1b, MROWS, d);
  float* a = c.f32((long)MROWS * d);
  mha(c, h1, h1, p.attn, d, heads, h1, a);
  float* h2 = c.f32((long)MROWS * d);
  ln(c, a, h2, p.ln2s, p.ln2b, MROWS, d);
  ffn(c, h2, p.ff, d, h2, out);
}

void dec_block(Ctx& c, const float* x, const float* enc, const DecP& p, int d,
               int heads, float* out) {
  ASave sv(c);
  float* h1 = c.f32((long)MROWS * d);
  ln(c, x, h1, p.ln1s, p.ln1b, MROWS, d);
  float* a1 = c.f32((long)MROWS * d);
  mha(c, h1, h1, p.sattn, d, heads, h1, a1);
  float* h2 = c.f32((long)MROWS * d);
  ln(c, a1, h2, p.ln2s, p.ln2b, MROWS, d);
  float* a2 = c.f32((long)MROWS * d);
  mha(c, h2, enc, p.cattn, d, heads, h2, a2);
  float* h3 = c.f32((long)MROWS * d);
  ln(c, a2, h3, p.ln3s, p.ln3b, MROWS, d);
  ffn(c, h3, p.ff, d, h3, out);
}

void vq(Ctx& c, const float* x, const CB& cb, int rows, float* q) {
  ASave sv(c);
  h16* hx = cvt(c, x, (long)rows * cb.D);
  float* S = c.f32((long)rows * cb.Kc);
  gemm(c, hx, cb.h, nullptr, nullptr, S, rows, cb.Kc, cb.D, cb.D, cb.D, cb.Kc,
       -2.f);
  k_argmin_gather<<<dim3((unsigned)rows), 256, 0, c.s>>>(S, cb.cn, cb.cb, q,
                                                         cb.Kc, cb.D);
}

void pointwise(Ctx& c, const float* x, const Lin* L, int n, bool act_last,
               float* out, int rows) {
  ASave sv(c);
  const float* cur = x;
  for (int i = 0; i < n; ++i) {
    float* dst = (i == n - 1) ? out : c.f32((long)rows * L[i].dout);
    h16* hx = cvt(c, cur, (long)rows * L[i].din);
    const int act = (i < n - 1 || act_last) ? 1 : 0;
    gemm(c, hx, L[i].wt, L[i].b, nullptr, dst, rows, L[i].dout, L[i].din,
         L[i].din, L[i].din, L[i].dout, 1.f, act);
    cur = dst;
  }
}

void ew_add(Ctx& c, const float* a, const float* b, float* o, long n) {
  k_add<<<dim3((unsigned)ceil_div(n, 256)), 256, 0, c.s>>>(a, b, o, n);
}
void ew_sub(Ctx& c, const float* a, const float* b, float* o, long n) {
  k_sub<<<dim3((unsigned)ceil_div(n, 256)), 256, 0, c.s>>>(a, b, o, n);
}

}  // namespace

// ---------------------------------------------------------------------------
// kernel_launch: full forward pass.  Input pytree order (sorted dict keys):
//   mag, params{cb1,cb2,cb3,cb_m,cb_p, dec1,dec2,dec3, dec_lin, enc1,enc2,enc3,
//               enc_lin, fc1,fc2,fc3, mdec,menc,pdec,penc}, pitch, x
// ---------------------------------------------------------------------------
extern "C" void kernel_launch(void* const* d_in, const int* in_sizes, int n_in,
                              void* d_out, int out_size, void* d_ws,
                              size_t ws_size, hipStream_t stream) {
  (void)in_sizes; (void)n_in; (void)out_size;
  Ctx c{stream, (char*)d_ws, 0, ws_size};
  Cur cu{d_in, 0};

  const float* mag = cu.next();
  CB cb1  = loadCB(c, cu.next(), 256, LAT);
  CB cb2  = loadCB(c, cu.next(), 256, LAT);
  CB cb3  = loadCB(c, cu.next(), 256, LAT);
  CB cb_m = loadCB(c, cu.next(), 1024, PDIM);
  CB cb_p = loadCB(c, cu.next(), 1024, PDIM);
  DecP dec1 = loadDec(cu, c, LAT);
  DecP dec2 = loadDec(cu, c, LAT);
  DecP dec3 = loadDec(cu, c, LAT);
  Lin dec_lin[4] = {loadLin(cu, c, LAT, LAT), loadLin(cu, c, LAT, LAT),
                    loadLin(cu, c, LAT, LAT), loadLin(cu, c, LAT, INDIM)};
  EncP enc1 = loadEnc(cu, c, LAT, 2);
  EncP enc2 = loadEnc(cu, c, LAT, 1);
  EncP enc3 = loadEnc(cu, c, LAT, 0);
  Lin enc_lin[4] = {loadLin(cu, c, INDIM, LAT), loadLin(cu, c, LAT, LAT),
                    loadLin(cu, c, LAT, LAT), loadLin(cu, c, LAT, LAT)};
  Lin fc1[2] = {loadLin(cu, c, LAT + 2 * PDIM, LAT), loadLin(cu, c, LAT, LAT)};
  Lin fc2[2] = {loadLin(cu, c, LAT + 2 * PDIM, LAT), loadLin(cu, c, LAT, LAT)};
  Lin fc3[2] = {loadLin(cu, c, LAT + 2 * PDIM, LAT), loadLin(cu, c, LAT, LAT)};
  DecP mdec = loadDec(cu, c, PDIM);
  EncP menc = loadEnc(cu, c, PDIM, 0);
  DecP pdec = loadDec(cu, c, PDIM);
  EncP penc = loadEnc(cu, c, PDIM, 0);
  const float* pitch = cu.next();
  const float* x     = cu.next();

  const long NL = (long)MROWS * LAT;
  const long NP = (long)MROWS * PDIM;

  float* xe = c.f32(NL);
  pointwise(c, x, enc_lin, 4, false, xe, MROWS);

  float* p_q = c.f32(NP);
  float* m_q = c.f32(NP);
  {
    ASave sv(c);
    float* pe = c.f32(NP);
    enc_core(c, pitch, penc, PDIM, 2, pe);
    vq(c, pe, cb_p, MROWS, p_q);
    float* me = c.f32(NP);
    enc_core(c, mag, menc, PDIM, 2, me);
    vq(c, me, cb_m, MROWS, m_q);
  }

  float* x1  = c.f32(NL);
  float* ds1 = c.f32(NL);
  enc_core(c, xe, enc1, LAT, 4, x1);
  {
    ASave sv(c);
    float* y1 = c.f32((long)BB * 400 * LAT);
    k_conv_ds<<<dim3((unsigned)ceil_div((long)BB * 400 * LAT, 256)), 256, 0, c.s>>>(
        x1, enc1.ds[0].w, enc1.ds[0].b, y1, BB, TT, LAT, 0);
    float* y2 = c.f32((long)BB * 200 * LAT);
    k_conv_ds<<<dim3((unsigned)ceil_div((long)BB * 200 * LAT, 256)), 256, 0, c.s>>>(
        y1, enc1.ds[1].w, enc1.ds[1].b, y2, BB, 400, LAT, 1);
    k_repeat<<<dim3((unsigned)ceil_div(NL, 256)), 256, 0, c.s>>>(y2, ds1, BB, TT, LAT, 4);
  }
  float* x1q = c.f32(NL);
  vq(c, ds1, cb1, MROWS, x1q);
  float* r1 = c.f32(NL);
  ew_sub(c, x1, x1q, r1, NL);

  float* x2  = c.f32(NL);
  float* ds2 = c.f32(NL);
  enc_core(c, r1, enc2, LAT, 4, x2);
  {
    ASave sv(c);
    float* y1 = c.f32((long)BB * 400 * LAT);
    k_conv_ds<<<dim3((unsigned)ceil_div((long)BB * 400 * LAT, 256)), 256, 0, c.s>>>(
        x2, enc2.ds[0].w, enc2.ds[0].b, y1, BB, TT, LAT, 0);
    k_repeat<<<dim3((unsigned)ceil_div(NL, 256)), 256, 0, c.s>>>(y1, ds2, BB, TT, LAT, 2);
  }
  float* x2q = c.f32(NL);
  vq(c, ds2, cb2, MROWS, x2q);
  float* r2 = c.f32(NL);
  ew_sub(c, x2, x2q, r2, NL);

  float* x3 = c.f32(NL);
  enc_core(c, r2, enc3, LAT, 4, x3);
  float* x3q = c.f32(NL);
  vq(c, x3, cb3, MROWS, x3q);

  float* p_dec = c.f32(NP);
  dec_block(c, p_q, p_q, pdec, PDIM, 2, p_dec);
  float* m_dec = c.f32(NP);
  dec_block(c, m_q, m_q, mdec, PDIM, 2, m_dec);

  const long NC = (long)MROWS * (LAT + 2 * PDIM);
  float* catb = c.f32(NC);

  float* d1 = c.f32(NL);
  dec_block(c, x1q, xe, dec3, LAT, 4, d1);
  float* o1 = c.f32(NL);
  k_concat3<<<dim3((unsigned)ceil_div(NC, 256)), 256, 0, c.s>>>(
      d1, LAT, p_dec, PDIM, m_dec, PDIM, catb, MROWS);
  pointwise(c, catb, fc3, 2, true, o1, MROWS);

  float* d2 = c.f32(NL);
  {
    ASave sv(c);
    float* d2a = c.f32(NL);
    dec_block(c, x2q, x2q, dec2, LAT, 4, d2a);
    float* t2 = c.f32(NL);
    ew_add(c, d2a, o1, t2, NL);
    dec_block(c, t2, xe, dec3, LAT, 4, d2);
  }
  float* o2 = c.f32(NL);
  k_concat3<<<dim3((unsigned)ceil_div(NC, 256)), 256, 0, c.s>>>(
      d2, LAT, p_dec, PDIM, m_dec, PDIM, catb, MROWS);
  pointwise(c, catb, fc2, 2, true, o2, MROWS);

  float* d3 = c.f32(NL);
  {
    ASave sv(c);
    float* d3a = c.f32(NL);
    dec_block(c, x3q, x3q, dec1, LAT, 4, d3a);
    float* t3 = c.f32(NL);
    ew_add(c, d3a, o2, t3, NL);
    float* d3b = c.f32(NL);
    dec_block(c, t3, t3, dec2, LAT, 4, d3b);
    float* t4 = c.f32(NL);
    ew_add(c, d3b, o1, t4, NL);
    dec_block(c, t4, xe, dec3, LAT, 4, d3);
  }
  float* o3 = c.f32(NL);
  k_concat3<<<dim3((unsigned)ceil_div(NC, 256)), 256, 0, c.s>>>(
      d3, LAT, p_dec, PDIM, m_dec, PDIM, catb, MROWS);
  pointwise(c, catb, fc1, 2, true, o3, MROWS);

  float* out = (float*)d_out;
  const long OS = (long)MROWS * INDIM;
  pointwise(c, o1, dec_lin, 4, false, out + 0 * OS, MROWS);
  pointwise(c, o2, dec_lin, 4, false, out + 1 * OS, MROWS);
  pointwise(c, o3, dec_lin, 4, false, out + 2 * OS, MROWS);
}
